// EncoderBlock_39548058862123
// MI455X (gfx1250) — matrix operand order repaired and checked by reading, not compiled, not run on hardware
//
#include <hip/hip_runtime.h>
#include <hip/hip_bf16.h>
#include <math.h>

// ---- problem constants (match reference) ----
#define BB   8
#define SS   1024
#define EE   1024
#define HH   16
#define DKK  64
#define FFF  4096
#define NTOK (BB * SS)

#define AS1 __attribute__((address_space(1)))
#define AS3 __attribute__((address_space(3)))

typedef __attribute__((ext_vector_type(16))) __bf16 bf16x16;
typedef __attribute__((ext_vector_type(8)))  float  f32x8;
typedef int v4i __attribute__((vector_size(16)));

union FragBF { bf16x16 v; uint4 u[2]; };   // 32-byte WMMA operand

__device__ __forceinline__ __bf16 f2bf(float f) {
    union { float f; unsigned u; } v; v.f = f;
    unsigned r = v.u + 0x7FFFu + ((v.u >> 16) & 1u);   // round-to-nearest-even
    unsigned short h = (unsigned short)(r >> 16);
    union { unsigned short s; __bf16 b; } o; o.s = h;
    return o.b;
}

// ---- async direct-to-LDS copy (CDNA5 GLOBAL_LOAD_ASYNC_TO_LDS_B128, ASYNCcnt) ----
#if __has_builtin(__builtin_amdgcn_global_load_async_to_lds_b128)
#define HAS_ASYNC 1
__device__ __forceinline__ void async_copy16(const __bf16* g, __bf16* l) {
    __builtin_amdgcn_global_load_async_to_lds_b128((AS1 v4i*)g, (AS3 v4i*)l, 0, 0);
}
__device__ __forceinline__ void async_wait0() {
#if __has_builtin(__builtin_amdgcn_s_wait_asynccnt)
    __builtin_amdgcn_s_wait_asynccnt(0);
#else
    asm volatile("s_wait_asynccnt 0x0" ::: "memory");
#endif
}
#else
#define HAS_ASYNC 0
#endif

// ---- 16-lane xor butterfly via v_permlane16_b32 (no LDS round trip) ----
#if __has_builtin(__builtin_amdgcn_permlane16)
#define HAS_PERMLANE 1
#else
#define HAS_PERMLANE 0
#endif

__device__ __forceinline__ float xor16(float x, int m, unsigned s0, unsigned s1) {
#if HAS_PERMLANE
    union { float f; unsigned u; } a; a.f = x;
    a.u = __builtin_amdgcn_permlane16(a.u, a.u, s0, s1, false, false);
    return a.f;
#else
    return __shfl_xor(x, m, 32);
#endif
}
__device__ __forceinline__ float rmax16(float x) {
    x = fmaxf(x, xor16(x, 1, 0x67452301u, 0xEFCDAB89u));
    x = fmaxf(x, xor16(x, 2, 0x54761032u, 0xDCFE98BAu));
    x = fmaxf(x, xor16(x, 4, 0x32107654u, 0xBA98FEDCu));
    x = fmaxf(x, xor16(x, 8, 0xFEDCBA98u, 0x76543210u));
    return x;
}
__device__ __forceinline__ float rsum16(float x) {
    x += xor16(x, 1, 0x67452301u, 0xEFCDAB89u);
    x += xor16(x, 2, 0x54761032u, 0xDCFE98BAu);
    x += xor16(x, 4, 0x32107654u, 0xBA98FEDCu);
    x += xor16(x, 8, 0xFEDCBA98u, 0x76543210u);
    return x;
}

// ---------------- fp32 [K][N] -> bf16 [N][K] transpose-cast (LDS tiled) ----------------
__global__ void k_cast_transpose(const float* __restrict__ in, __bf16* __restrict__ out,
                                 int K, int N) {
    __shared__ float tile[32][33];
    int kb = blockIdx.y * 32, nb = blockIdx.x * 32;
    int tx = threadIdx.x, ty = threadIdx.y;            // 32 x 8
#pragma unroll
    for (int i = 0; i < 32; i += 8)
        tile[ty + i][tx] = in[(size_t)(kb + ty + i) * N + nb + tx];
    __syncthreads();
#pragma unroll
    for (int i = 0; i < 32; i += 8)
        out[(size_t)(nb + ty + i) * K + kb + tx] = f2bf(tile[tx][ty + i]);
}

// ---------------- RMSNorm (fp32 in, bf16 out) ----------------
__global__ void k_rmsnorm(const float* __restrict__ x, const float* __restrict__ g,
                          __bf16* __restrict__ out) {
    __shared__ float red[8];
    int row = blockIdx.x;
    int t = threadIdx.x;                               // 256 threads, 8 waves
    const float* xr = x + (size_t)row * EE;
    float vals[4];
    float ss = 0.f;
#pragma unroll
    for (int i = 0; i < 4; ++i) { vals[i] = xr[t + i * 256]; ss += vals[i] * vals[i]; }
    ss = rsum16(ss);
    ss += __shfl_xor(ss, 16, 32);                      // fold the two 16-lane halves
    if ((t & 31) == 0) red[t >> 5] = ss;
    __syncthreads();
    float tot = 0.f;
#pragma unroll
    for (int i = 0; i < 8; ++i) tot += red[i];
    float rs = rsqrtf(tot / (float)EE + 1.1920929e-07f);
#pragma unroll
    for (int i = 0; i < 4; ++i)
        out[(size_t)row * EE + t + i * 256] = f2bf(vals[i] * rs * g[t + i * 256]);
}

// ---------------- GEMM tile staging (async-to-LDS when available) ----------------
__device__ __forceinline__ void stage_gemm(const __bf16* __restrict__ A,
                                           const __bf16* __restrict__ Bt,
                                           int m0, int n0, int K, int k0, int t,
                                           __bf16 (*lA)[32], __bf16 (*lBt)[32]) {
#pragma unroll
    for (int r = 0; r < 2; ++r) {                      // A tile 128x32 = 512 x 16B
        int idx = t + r * 256;
        int row = idx >> 2, seg = idx & 3;
        const __bf16* src = A + (size_t)(m0 + row) * K + k0 + seg * 8;
#if HAS_ASYNC
        async_copy16(src, &lA[row][seg * 8]);
#else
        *(uint4*)&lA[row][seg * 8] = *(const uint4*)src;
#endif
    }
    {                                                  // Bt tile 64x32 = 256 x 16B
        int row = t >> 2, seg = t & 3;
        const __bf16* src = Bt + (size_t)(n0 + row) * K + k0 + seg * 8;
#if HAS_ASYNC
        async_copy16(src, &lBt[row][seg * 8]);
#else
        *(uint4*)&lBt[row][seg * 8] = *(const uint4*)src;
#endif
    }
}

// ---------------- tiled WMMA GEMM: C = A(bf16,[M][K]) * Bt(bf16,[N][K])^T + bias ----------------
// BM=128, BN=64, KB=32; 256 threads = 8 waves; double-buffered LDS, async pipeline.
template <bool RELU, bool BF16OUT>
__global__ void k_gemm(const __bf16* __restrict__ A, const __bf16* __restrict__ Bt,
                       const float* __restrict__ bias, const float* __restrict__ resid,
                       void* __restrict__ outp, int M, int N, int K) {
    __shared__ __align__(32) __bf16 lA[2][128][32];    // [buf][m][k]
    __shared__ __align__(32) __bf16 lBt[2][64][32];    // [buf][n][k]
    int t = threadIdx.x;
    int wave = t >> 5, lane = t & 31;
    int lhalf = lane >> 4, lmod = lane & 15;
    int m0 = blockIdx.y * 128, n0 = blockIdx.x * 64;
    int mrow = wave * 16 + lmod;

    f32x8 acc[4] = {};

    stage_gemm(A, Bt, m0, n0, K, 0, t, lA[0], lBt[0]);
#if HAS_ASYNC
    async_wait0();
#endif
    __syncthreads();

    int buf = 0;
    for (int k0 = 0; k0 < K; k0 += 32) {
        if (k0 + 32 < K) {                             // prefetch next tile into other buffer
            stage_gemm(A, Bt, m0, n0, K, k0 + 32, t, lA[buf ^ 1], lBt[buf ^ 1]);
        }
#if !HAS_ASYNC
        else {
            __builtin_prefetch(Bt + (size_t)(n0 + (t >> 2)) * K + k0, 0, 1);
        }
#endif
        // A fragment: two contiguous 16B halves (k = 16*lhalf + s, same order as B)
        FragBF af;
        af.u[0] = *(const uint4*)&lA[buf][mrow][lhalf * 16];
        af.u[1] = *(const uint4*)&lA[buf][mrow][lhalf * 16 + 8];
#pragma unroll
        for (int j = 0; j < 4; ++j) {
            // B fragment: 16 contiguous k per lane -> one 32B LDS read
            bf16x16 bfr = *(const bf16x16*)&lBt[buf][j * 16 + lmod][lhalf * 16];
            acc[j] = __builtin_amdgcn_wmma_f32_16x16x32_bf16(
                false, af.v, false, bfr, (short)0, acc[j], false, false);
        }
#if HAS_ASYNC
        async_wait0();                                 // next tile landed in LDS
#endif
        __syncthreads();
        buf ^= 1;
    }
    // epilogue: C layout M = v + 8*lhalf, N = lmod
#pragma unroll
    for (int j = 0; j < 4; ++j) {
#pragma unroll
        for (int v = 0; v < 8; ++v) {
            int m = m0 + wave * 16 + v + lhalf * 8;
            int n = n0 + j * 16 + lmod;
            float val = acc[j][v] + bias[n];
            if (resid) val += resid[(size_t)m * N + n];
            if (RELU) val = fmaxf(val, 0.f);
            if (BF16OUT) ((__bf16*)outp)[(size_t)m * N + n] = f2bf(val);
            else         ((float*)outp)[(size_t)m * N + n] = val;
        }
    }
}

// ---------------- RoPE + head-major relayout (fp32 -> bf16) ----------------
// Q,K -> [B,H,S,DK] (k-dim = d contiguous); V -> [B,H,DK,S] (V^T, k-dim = key contiguous).
__global__ void k_rope(const float* __restrict__ Qf, const float* __restrict__ Kf,
                       const float* __restrict__ Vf,
                       __bf16* __restrict__ qbf, __bf16* __restrict__ kbf,
                       __bf16* __restrict__ vtbf) {
    int d = threadIdx.x;                               // 0..63
    int s = blockIdx.x, h = blockIdx.y, b = blockIdx.z;
    size_t tok = (size_t)(b * SS + s);
    size_t src = tok * EE + h * DKK + d;
    float q = Qf[src], k = Kf[src], v = Vf[src];
    int i = d & 31;
    float freq = __expf(-((float)(2 * i) / (float)DKK) * 9.210340371976184f);
    float ang = (float)s * freq;
    float c = __cosf(ang), sn = __sinf(ang);
    size_t src2 = tok * EE + h * DKK + ((d < 32) ? d + 32 : d - 32);
    float qr = (d < 32) ? -Qf[src2] : Qf[src2];
    float kr = (d < 32) ? -Kf[src2] : Kf[src2];
    size_t bh = (size_t)(b * HH + h);
    qbf[(bh * SS + s) * DKK + d]  = f2bf(q * c + qr * sn);
    kbf[(bh * SS + s) * DKK + d]  = f2bf(k * c + kr * sn);
    vtbf[(bh * DKK + d) * SS + s] = f2bf(v);
}

// ---------------- flash attention: 128 threads = 4 waves, 4 query tiles / block ----------------
__global__ void k_attn(const __bf16* __restrict__ qbf, const __bf16* __restrict__ kbf,
                       const __bf16* __restrict__ vtbf, __bf16* __restrict__ obf) {
    __shared__ __align__(32) __bf16 kT[32][64];        // [key][d]   (score B fragments)
    __shared__ __align__(32) __bf16 vT[64][32];        // [d][key]   (PV    B fragments)
    __shared__ __align__(32) __bf16 lP[4][16][32];     // per-wave P tile [q][key]
    int t = threadIdx.x;                               // 128
    int wave = t >> 5, lane = t & 31;
    int lhalf = lane >> 4, lmod = lane & 15;
    int q0 = blockIdx.x * 64 + wave * 16;
    int h = blockIdx.y, b = blockIdx.z;
    size_t bh = (size_t)(b * HH + h);

    // preload Q fragments for d-chunks {0..31, 32..63}: vectorized 16B halves
    FragBF qa[2];
    {
        const __bf16* qrow = qbf + (bh * SS + q0 + lmod) * DKK;
#pragma unroll
        for (int ch = 0; ch < 2; ++ch) {
            qa[ch].u[0] = *(const uint4*)(qrow + ch * 32 + lhalf * 16);
            qa[ch].u[1] = *(const uint4*)(qrow + ch * 32 + lhalf * 16 + 8);
        }
    }

    f32x8 o[4] = {};
    float rowm[8], rowl[8];
#pragma unroll
    for (int v = 0; v < 8; ++v) { rowm[v] = -__builtin_inff(); rowl[v] = 0.f; }
    const float scale = 0.125f;                        // 1/sqrt(64)

    for (int kt = 0; kt < SS; kt += 32) {
        // stage K tile [32 keys][64 d] + V^T tile [64 d][32 keys] (async-to-LDS if available)
#pragma unroll
        for (int r = 0; r < 2; ++r) {
            int idx = t + r * 128;
            int krow = idx >> 3, kseg = idx & 7;
            const __bf16* ksrc = kbf + (bh * SS + kt + krow) * DKK + kseg * 8;
            int vrow = idx >> 2, vseg = idx & 3;
            const __bf16* vsrc = vtbf + (bh * DKK + vrow) * SS + kt + vseg * 8;
#if HAS_ASYNC
            async_copy16(ksrc, &kT[krow][kseg * 8]);
            async_copy16(vsrc, &vT[vrow][vseg * 8]);
#else
            *(uint4*)&kT[krow][kseg * 8] = *(const uint4*)ksrc;
            *(uint4*)&vT[vrow][vseg * 8] = *(const uint4*)vsrc;
#endif
        }
#if HAS_ASYNC
        async_wait0();
#endif
        __syncthreads();

        // scores for two 16-key tiles, accumulated over d (2 chunks of 32)
        f32x8 c0 = {}, c1 = {};
#pragma unroll
        for (int ch = 0; ch < 2; ++ch) {
            bf16x16 kb0 = *(const bf16x16*)&kT[lmod][ch * 32 + lhalf * 16];
            bf16x16 kb1 = *(const bf16x16*)&kT[16 + lmod][ch * 32 + lhalf * 16];
            c0 = __builtin_amdgcn_wmma_f32_16x16x32_bf16(false, qa[ch].v, false, kb0, (short)0, c0, false, false);
            c1 = __builtin_amdgcn_wmma_f32_16x16x32_bf16(false, qa[ch].v, false, kb1, (short)0, c1, false, false);
        }
        // online softmax; row M = v + 8*lhalf lives across 16 lanes of this half
#pragma unroll
        for (int v = 0; v < 8; ++v) {
            float s0 = c0[v] * scale, s1 = c1[v] * scale;
            float tmax = rmax16(fmaxf(s0, s1));
            float mnew = fmaxf(rowm[v], tmax);
            float p0 = __expf(s0 - mnew), p1 = __expf(s1 - mnew);
            float ps = rsum16(p0 + p1);
            float f = __expf(rowm[v] - mnew);
            rowl[v] = rowl[v] * f + ps;
            rowm[v] = mnew;
#pragma unroll
            for (int j = 0; j < 4; ++j) o[j][v] *= f;
            int r = v + lhalf * 8;
            lP[wave][r][lmod]      = f2bf(p0);
            lP[wave][r][16 + lmod] = f2bf(p1);
        }
        __syncthreads();                               // lP visible (kT reads already done)

        // P as A-fragment (16 q x 32 keys): two 16B LDS reads
        FragBF pa;
        pa.u[0] = *(const uint4*)&lP[wave][lmod][lhalf * 16];
        pa.u[1] = *(const uint4*)&lP[wave][lmod][lhalf * 16 + 8];
        // PV: 4 output tiles along d; V^T fragment is one 32B LDS read
#pragma unroll
        for (int j = 0; j < 4; ++j) {
            bf16x16 vb = *(const bf16x16*)&vT[j * 16 + lmod][lhalf * 16];
            o[j] = __builtin_amdgcn_wmma_f32_16x16x32_bf16(false, pa.v, false, vb, (short)0, o[j], false, false);
        }
        __syncthreads();                               // kT/vT free for next staging
    }
    // normalize + store into [token][E] bf16 for the O-projection GEMM
#pragma unroll
    for (int j = 0; j < 4; ++j)
#pragma unroll
        for (int v = 0; v < 8; ++v) {
            int m = v + lhalf * 8;
            float val = o[j][v] / rowl[v];
            obf[((size_t)(b * SS) + q0 + m) * EE + h * DKK + j * 16 + lmod] = f2bf(val);
        }
}

// ---------------- host orchestration ----------------
extern "C" void kernel_launch(void* const* d_in, const int* in_sizes, int n_in,
                              void* d_out, int out_size, void* d_ws, size_t ws_size,
                              hipStream_t stream) {
    const float* x  = (const float*)d_in[0];
    const float* WQ = (const float*)d_in[1];  const float* bQ = (const float*)d_in[2];
    const float* WK = (const float*)d_in[3];  const float* bK = (const float*)d_in[4];
    const float* WV = (const float*)d_in[5];  const float* bV = (const float*)d_in[6];
    const float* WO = (const float*)d_in[7];  const float* bO = (const float*)d_in[8];
    const float* W1 = (const float*)d_in[9];  const float* b1 = (const float*)d_in[10];
    const float* W2 = (const float*)d_in[11]; const float* b2 = (const float*)d_in[12];
    const float* g1 = (const float*)d_in[13]; const float* g2 = (const float*)d_in[14];

    const size_t MB = 1ull << 20;
    char* ws = (char*)d_ws;
    __bf16* wqT = (__bf16*)(ws + 0 * MB);     // [N][K] transposed bf16 weights
    __bf16* wkT = (__bf16*)(ws + 2 * MB);
    __bf16* wvT = (__bf16*)(ws + 4 * MB);
    __bf16* woT = (__bf16*)(ws + 6 * MB);
    __bf16* w1T = (__bf16*)(ws + 8 * MB);     // [FF][E]
    __bf16* w2T = (__bf16*)(ws + 16 * MB);    // [E][FF]
    __bf16* hbf = (__bf16*)(ws + 24 * MB);    // normed activations (reused twice)
    float*  x2  = (float*) (ws + 40 * MB);    // attention residual output
    __bf16* qbf = (__bf16*)(ws + 72 * MB);    // [B,H,S,DK]
    __bf16* kbf = (__bf16*)(ws + 88 * MB);    // [B,H,S,DK]
    __bf16* vtbf= (__bf16*)(ws + 104 * MB);   // [B,H,DK,S]
    __bf16* obf = (__bf16*)(ws + 120 * MB);   // [token][E]
    float*  Qf  = (float*) (ws + 136 * MB);
    float*  Kf  = (float*) (ws + 168 * MB);
    float*  Vf  = (float*) (ws + 200 * MB);
    __bf16* f1  = (__bf16*)(ws + 136 * MB);   // reuses Qf/Kf space after RoPE

    dim3 tb(32, 8);
    // 1) weights -> bf16, transposed to [N][K]
    k_cast_transpose<<<dim3(EE / 32, EE / 32),  tb, 0, stream>>>(WQ, wqT, EE, EE);
    k_cast_transpose<<<dim3(EE / 32, EE / 32),  tb, 0, stream>>>(WK, wkT, EE, EE);
    k_cast_transpose<<<dim3(EE / 32, EE / 32),  tb, 0, stream>>>(WV, wvT, EE, EE);
    k_cast_transpose<<<dim3(EE / 32, EE / 32),  tb, 0, stream>>>(WO, woT, EE, EE);
    k_cast_transpose<<<dim3(FFF / 32, EE / 32), tb, 0, stream>>>(W1, w1T, EE, FFF);
    k_cast_transpose<<<dim3(EE / 32, FFF / 32), tb, 0, stream>>>(W2, w2T, FFF, EE);

    // 2) h1 = rmsnorm(x, g1)
    k_rmsnorm<<<NTOK, 256, 0, stream>>>(x, g1, hbf);

    // 3) Q/K/V projections (fp32 out, bias, no residual)
    dim3 gE(EE / 64, NTOK / 128);
    k_gemm<false, false><<<gE, 256, 0, stream>>>(hbf, wqT, bQ, nullptr, Qf, NTOK, EE, EE);
    k_gemm<false, false><<<gE, 256, 0, stream>>>(hbf, wkT, bK, nullptr, Kf, NTOK, EE, EE);
    k_gemm<false, false><<<gE, 256, 0, stream>>>(hbf, wvT, bV, nullptr, Vf, NTOK, EE, EE);

    // 4) RoPE + relayout to bf16 (Q,K head-major; V transposed)
    k_rope<<<dim3(SS, HH, BB), DKK, 0, stream>>>(Qf, Kf, Vf, qbf, kbf, vtbf);

    // 5) flash attention
    k_attn<<<dim3(SS / 64, HH, BB), 128, 0, stream>>>(qbf, kbf, vtbf, obf);

    // 6) x2 = x + O @ WO + bO
    k_gemm<false, false><<<gE, 256, 0, stream>>>(obf, woT, bO, x, x2, NTOK, EE, EE);

    // 7) h2 = rmsnorm(x2, g2)
    k_rmsnorm<<<NTOK, 256, 0, stream>>>(x2, g2, hbf);

    // 8) f1 = relu(h2 @ W1 + b1)  (bf16 out)
    dim3 gF(FFF / 64, NTOK / 128);
    k_gemm<true, true><<<gF, 256, 0, stream>>>(hbf, w1T, b1, nullptr, f1, NTOK, FFF, EE);

    // 9) out = x2 + f1 @ W2 + b2  (fp32 out)
    k_gemm<false, false><<<gE, 256, 0, stream>>>(f1, w2T, b2, x2, (float*)d_out, NTOK, EE, FFF);
}
